// SpikeTimeSeriesForecaster_13486197309585
// MI455X (gfx1250) — matrix operand order repaired and checked by reading, compiled, not run on hardware
//
#include <hip/hip_runtime.h>

// ---------------------------------------------------------------------------
// Fused 3-layer LIF spiking network, persistent per-row-tile kernel.
// Each block owns 16 batch rows and runs GEMM1 + all 50 timesteps locally.
// f16 WMMA (v_wmma_f32_16x16x32_f16) with f32 accumulation / f32 LIF state.
// ---------------------------------------------------------------------------

typedef __attribute__((ext_vector_type(16))) _Float16 v16h;
typedef __attribute__((ext_vector_type(8)))  _Float16 v8h;
typedef __attribute__((ext_vector_type(8)))  float    v8f;

union U16 { v16h v; v8h h[2]; };

// Problem sizes (fixed by the reference)
constexpr int BATCH = 8192, IN = 256, H1 = 512, H2 = 256, NOUT = 64;
constexpr int T = 50, WIN = 10;
constexpr int ROWS = 16;                 // batch rows per block
constexpr float DECAY = 0.9f, THRESH = 1.0f, REFRAC = 2.0f;

// LDS strides in halves (padded to break bank alignment of row strides)
constexpr int LD_X  = 264;   // x tile  [16][256]
constexpr int LD_S1 = 520;   // s1      [16][512]
constexpr int LD_S2 = 264;   // s2      [16][256]
constexpr int LD_W1 = 260;   // W1^T    [512][256]
constexpr int LD_W2 = 516;   // W2^T    [256][512]
constexpr int LD_W3 = 264;   // W3^T    [64][256]

// LDS regions (bytes)
constexpr size_t OFF_S1 = 0;                         // also holds x tile (GEMM1 only)
constexpr size_t OFF_S2 = (size_t)ROWS * LD_S1 * 2;  // 16640
constexpr size_t OFF_W  = OFF_S2 + (size_t)ROWS * LD_S2 * 2;          // 25088
constexpr size_t OFF_W3 = OFF_W + (size_t)H1 * LD_W1 * 2;             // 291328
constexpr size_t SMEM_BYTES = OFF_W3 + (size_t)NOUT * LD_W3 * 2;      // 325120 <= 320KB

// --- fragment loaders (ISA 7.12.2 16-bit layouts) ---------------------------
// A (16x32, MxK): lane l: row = l&15; halves K = 8h..8h+7 and 16+8h..16+8h+7
__device__ __forceinline__ v16h load_a_frag(const _Float16* buf, int ld, int kc,
                                            int ln, int hh) {
  U16 u;
  const _Float16* p = buf + ln * ld + kc + 8 * hh;
  u.h[0] = *(const v8h*)(p);
  u.h[1] = *(const v8h*)(p + 16);
  return u.v;
}
// B (32x16, KxN) from column-major-in-LDS weights (Wt[n][k], k contiguous):
// lane l: col = ncol + (l&15); halves K = 8h..8h+7 and 16+8h..16+8h+7
__device__ __forceinline__ v16h load_b_frag(const _Float16* buf, int ld, int ncol,
                                            int kc, int ln, int hh) {
  U16 u;
  const _Float16* p = buf + (ncol + ln) * ld + kc + 8 * hh;
  u.h[0] = *(const v8h*)(p);
  u.h[1] = *(const v8h*)(p + 16);
  return u.v;
}

__device__ __forceinline__ v8f wmma16(v16h a, v16h b, v8f c) {
  // D = A x B + C, f32 accumulate
  return __builtin_amdgcn_wmma_f32_16x16x32_f16(false, a, false, b, (short)0, c,
                                                false, false);
}

// LIF update (macro so vector-element lvalues work without references)
#define LIF_STEP(Vv, Rr, Ii, Ss)                                 \
  {                                                              \
    float _V = DECAY * (Vv) + (Ii);                              \
    float _s = ((_V - THRESH >= 0.0f) && ((Rr) <= 0.0f)) ? 1.0f : 0.0f; \
    (Vv) = _V * (1.0f - _s);                                     \
    (Rr) = fmaxf((Rr) - 1.0f + REFRAC * _s, 0.0f);               \
    (Ss) = _s;                                                   \
  }

__global__ __launch_bounds__(256) void snn_fused_kernel(
    const float* __restrict__ x,  const float* __restrict__ W1,
    const float* __restrict__ b1, const float* __restrict__ W2,
    const float* __restrict__ b2, const float* __restrict__ W3,
    const float* __restrict__ b3, float* __restrict__ out) {
  extern __shared__ char smem[];
  _Float16* sA  = (_Float16*)(smem + OFF_S1);  // x tile, then s1
  _Float16* sS2 = (_Float16*)(smem + OFF_S2);  // s2
  _Float16* sW  = (_Float16*)(smem + OFF_W);   // W1^T, then W2^T
  _Float16* sW3 = (_Float16*)(smem + OFF_W3);  // W3^T

  const int tid  = threadIdx.x;
  const int wv   = tid >> 5;        // wave id 0..7
  const int lane = tid & 31;
  const int ln   = lane & 15;
  const int hh   = lane >> 4;
  const int rowBase = blockIdx.x * ROWS;

  // ---- stage x tile (f32->f16) and W1^T into LDS --------------------------
  for (int idx = tid; idx < ROWS * IN; idx += 256) {
    int r = idx >> 8, c = idx & (IN - 1);
    sA[r * LD_X + c] = (_Float16)x[(size_t)(rowBase + r) * IN + c];
  }
  for (int idx = tid; idx < IN * H1; idx += 256) {
    int k = idx >> 9, n = idx & (H1 - 1);      // W1 is [256][512] row-major
    sW[n * LD_W1 + k] = (_Float16)W1[idx];
  }
  __syncthreads();

  // ---- GEMM1: I1 = x @ W1 + b1 ; wave owns 4 N-tiles (64 cols) ------------
  v8f zf = {};
  v8f i1[4] = {zf, zf, zf, zf};
  for (int kc = 0; kc < IN; kc += 32) {
    v16h a = load_a_frag(sA, LD_X, kc, ln, hh);
#pragma unroll
    for (int tn = 0; tn < 4; ++tn) {
      v16h b = load_b_frag(sW, LD_W1, wv * 64 + tn * 16, kc, ln, hh);
      i1[tn] = wmma16(a, b, i1[tn]);
    }
  }
#pragma unroll
  for (int tn = 0; tn < 4; ++tn) {
    float bb = b1[wv * 64 + tn * 16 + ln];
#pragma unroll
    for (int i = 0; i < 8; ++i) i1[tn][i] += bb;
  }
  __syncthreads();  // everyone done with x / W1^T regions

  // ---- stage W2^T (overwrites W1^T region) and W3^T -----------------------
  for (int idx = tid; idx < H1 * H2; idx += 256) {
    int k = idx >> 8, n = idx & (H2 - 1);      // W2 is [512][256]
    sW[n * LD_W2 + k] = (_Float16)W2[idx];
  }
  for (int idx = tid; idx < H2 * NOUT; idx += 256) {
    int k = idx >> 6, n = idx & (NOUT - 1);    // W3 is [256][64]
    sW3[n * LD_W3 + k] = (_Float16)W3[idx];
  }

  // biases for layers 2/3 (b vectors are tiny; keep in regs)
  float bb2[2];
#pragma unroll
  for (int tn = 0; tn < 2; ++tn) bb2[tn] = b2[wv * 32 + tn * 16 + ln];
  float bb3 = (wv < 4) ? b3[wv * 16 + ln] : 0.0f;

  // ---- LIF state (all in fragment layout) ---------------------------------
  v8f v1[4] = {zf, zf, zf, zf}, r1[4] = {zf, zf, zf, zf};
  v8f v2[2] = {zf, zf}, r2[2] = {zf, zf};
  v8f v3 = zf, r3 = zf, acc = zf;

  __syncthreads();  // W2^T / W3^T visible

  // ---- timestep loop ------------------------------------------------------
  for (int t = 0; t < T; ++t) {
    // LIF layer 1 -> s1 (f16, LDS)
#pragma unroll
    for (int tn = 0; tn < 4; ++tn) {
      int col = wv * 64 + tn * 16 + ln;
#pragma unroll
      for (int i = 0; i < 8; ++i) {
        float sp;
        LIF_STEP(v1[tn][i], r1[tn][i], i1[tn][i], sp);
        sA[(i + 8 * hh) * LD_S1 + col] = (_Float16)sp;
      }
    }
    __syncthreads();  // s1 published

    // GEMM2: I2 = s1 @ W2 ; wave owns 2 N-tiles (32 cols)
    v8f i2[2] = {zf, zf};
    for (int kc = 0; kc < H1; kc += 32) {
      v16h a = load_a_frag(sA, LD_S1, kc, ln, hh);
#pragma unroll
      for (int tn = 0; tn < 2; ++tn) {
        v16h b = load_b_frag(sW, LD_W2, wv * 32 + tn * 16, kc, ln, hh);
        i2[tn] = wmma16(a, b, i2[tn]);
      }
    }
    // LIF layer 2 -> s2 (f16, LDS)
#pragma unroll
    for (int tn = 0; tn < 2; ++tn) {
      int col = wv * 32 + tn * 16 + ln;
#pragma unroll
      for (int i = 0; i < 8; ++i) {
        float I = i2[tn][i] + bb2[tn];
        float sp;
        LIF_STEP(v2[tn][i], r2[tn][i], I, sp);
        sS2[(i + 8 * hh) * LD_S2 + col] = (_Float16)sp;
      }
    }
    __syncthreads();  // s2 published; all s1 reads done

    // GEMM3 + LIF3 + rate accumulation: waves 0..3 own the 4 N-tiles of OUT
    if (wv < 4) {
      v8f i3 = zf;
      for (int kc = 0; kc < H2; kc += 32) {
        v16h a = load_a_frag(sS2, LD_S2, kc, ln, hh);
        v16h b = load_b_frag(sW3, LD_W3, wv * 16, kc, ln, hh);
        i3 = wmma16(a, b, i3);
      }
#pragma unroll
      for (int i = 0; i < 8; ++i) {
        float I = i3[i] + bb3;
        float sp;
        LIF_STEP(v3[i], r3[i], I, sp);
        if (t >= T - WIN) acc[i] += sp;
      }
    }
  }

  // ---- write rate-decoded output ------------------------------------------
  if (wv < 4) {
    int col = wv * 16 + ln;
#pragma unroll
    for (int i = 0; i < 8; ++i) {
      int row = rowBase + i + 8 * hh;
      out[(size_t)row * NOUT + col] = acc[i] * (1.0f / (float)WIN);
    }
  }
}

extern "C" void kernel_launch(void* const* d_in, const int* in_sizes, int n_in,
                              void* d_out, int out_size, void* d_ws, size_t ws_size,
                              hipStream_t stream) {
  const float* x  = (const float*)d_in[0];
  const float* W1 = (const float*)d_in[1];
  const float* b1 = (const float*)d_in[2];
  const float* W2 = (const float*)d_in[3];
  const float* b2 = (const float*)d_in[4];
  const float* W3 = (const float*)d_in[5];
  const float* b3 = (const float*)d_in[6];
  float* out = (float*)d_out;

  // Allow >64KB dynamic LDS (CDNA5 WGP supports 320KB per workgroup).
  (void)hipFuncSetAttribute((const void*)snn_fused_kernel,
                            hipFuncAttributeMaxDynamicSharedMemorySize,
                            (int)SMEM_BYTES);

  dim3 grid(BATCH / ROWS);  // 512 blocks, 16 rows each
  snn_fused_kernel<<<grid, 256, SMEM_BYTES, stream>>>(x, W1, b1, W2, b2, W3, b3,
                                                      out);
}
